// VectorQuantizer_66984309948643
// MI455X (gfx1250) — matrix operand order repaired and checked by reading, not compiled, not run on hardware
//
#include <hip/hip_runtime.h>

#define EMB 64
#define NCODES 1024
#define ROWS_PER_BLOCK 128
#define CHUNK 256        // codes staged into LDS per chunk
#define XPAD 72          // padded LDS row stride (f16 elems): 144B = 36 dwords -> conflict-free
#define CBPAD 72

typedef __attribute__((ext_vector_type(16))) _Float16 v16h;
typedef __attribute__((ext_vector_type(8)))  float    v8f;
typedef __attribute__((ext_vector_type(4)))  _Float16 v4h;

__global__ __launch_bounds__(256) void vq_argmin_kernel(
    const float* __restrict__ x,       // [N, 64] fp32
    const float* __restrict__ cb,      // [1024, 64] fp32
    float* __restrict__ out_q,         // [N, 64] fp32
    float* __restrict__ out_idx)       // [N] as float
{
    __shared__ _Float16 lds_x[ROWS_PER_BLOCK * XPAD];   // 18432 B
    __shared__ _Float16 lds_cb[CHUNK * CBPAD];          // 36864 B  (holds -2*w in f16)
    __shared__ float    lds_norm[NCODES];               //  4096 B
    __shared__ int      lds_idx[ROWS_PER_BLOCK];        //   512 B

    const int tid  = threadIdx.x;
    const int wave = tid >> 5;
    const int lane = tid & 31;
    const int hgrp = lane >> 4;      // which 16-lane half
    const int lcol = lane & 15;

    const long rowbase = (long)blockIdx.x * ROWS_PER_BLOCK;

    // ---- stage x tile (fp32 -> f16, coalesced float4 reads) ----
    #pragma unroll
    for (int j = 0; j < 8; ++j) {
        int i4  = tid + 256 * j;             // 2048 float4 total
        int row = i4 >> 4;
        int c4  = i4 & 15;
        float4 v = ((const float4*)x)[(rowbase + row) * (EMB / 4) + c4];
        v4h h;
        h.x = (_Float16)v.x; h.y = (_Float16)v.y;
        h.z = (_Float16)v.z; h.w = (_Float16)v.w;
        *(v4h*)&lds_x[row * XPAD + c4 * 4] = h;
    }

    // ---- codebook squared norms in fp32 (codebook is L2-resident) ----
    #pragma unroll
    for (int j = 0; j < 4; ++j) {
        int code = tid + 256 * j;
        const float4* p = (const float4*)(cb + code * EMB);
        float s = 0.0f;
        #pragma unroll
        for (int q = 0; q < EMB / 4; ++q) {
            float4 v = p[q];
            s += v.x * v.x + v.y * v.y + v.z * v.z + v.w * v.w;
        }
        lds_norm[code] = s;
    }
    __syncthreads();

    // ---- build A fragments once (this wave's 16 rows, K=0..63) ----
    // ISA 16-bit A 16x32 layout: lanes 0-15 hold K={8h.., 16+8h..} with h=lane/16
    const int mrow = wave * 16 + lcol;
    union AF { v16h v; uint4 q[2]; } ua0, ua1;
    ua0.q[0] = *(const uint4*)&lds_x[mrow * XPAD +      8 * hgrp];  // K =  8h+0..7
    ua0.q[1] = *(const uint4*)&lds_x[mrow * XPAD + 16 + 8 * hgrp];  // K = 16+8h+..
    ua1.q[0] = *(const uint4*)&lds_x[mrow * XPAD + 32 + 8 * hgrp];  // K = 32+8h+..
    ua1.q[1] = *(const uint4*)&lds_x[mrow * XPAD + 48 + 8 * hgrp];  // K = 48+8h+..

    float best[8];
    int   bidx[8];
    #pragma unroll
    for (int r = 0; r < 8; ++r) { best[r] = 3.4e38f; bidx[r] = 0; }

    // ---- sweep codebook in chunks of 256 codes ----
    for (int ch = 0; ch < NCODES / CHUNK; ++ch) {
        __syncthreads();   // previous chunk fully consumed
        #pragma unroll
        for (int j = 0; j < 16; ++j) {
            int i4   = tid + 256 * j;        // 4096 float4 per chunk
            int code = i4 >> 4;
            int c4   = i4 & 15;
            float4 v = ((const float4*)cb)[(ch * CHUNK + code) * (EMB / 4) + c4];
            v4h h;                            // stage -2*w so WMMA emits -2*x.w directly
            h.x = (_Float16)(-2.0f * v.x); h.y = (_Float16)(-2.0f * v.y);
            h.z = (_Float16)(-2.0f * v.z); h.w = (_Float16)(-2.0f * v.w);
            *(v4h*)&lds_cb[code * CBPAD + c4 * 4] = h;
        }
        // prefetch next chunk's fp32 codebook lines into cache while we compute
        if (ch + 1 < NCODES / CHUNK) {
            const char* nxt = (const char*)(cb + (ch + 1) * CHUNK * EMB);
            __builtin_prefetch(nxt + tid * 256, 0, 3);
        }
        __syncthreads();

        for (int nt = 0; nt < CHUNK / 16; ++nt) {
            // B 32x16: lane holds column (code) lcol, K = {8h+0..7, 16+8h+0..7} (matches A)
            const int crow = nt * 16 + lcol;
            union BF { v16h v; uint4 q[2]; } ub0, ub1;
            ub0.q[0] = *(const uint4*)&lds_cb[crow * CBPAD + 8 * hgrp];
            ub0.q[1] = *(const uint4*)&lds_cb[crow * CBPAD + 16 + 8 * hgrp];
            ub1.q[0] = *(const uint4*)&lds_cb[crow * CBPAD + 32 + 8 * hgrp];
            ub1.q[1] = *(const uint4*)&lds_cb[crow * CBPAD + 48 + 8 * hgrp];

            const int   code = ch * CHUNK + nt * 16 + lcol;   // this lane's column
            const float nrm  = lds_norm[code];

            // Seed C with ||w||^2 (per-lane constant across the 8 rows); after the
            // two WMMAs acc[r] == ||w||^2 - 2*x.w  (the comparison score directly).
            v8f acc = { nrm, nrm, nrm, nrm, nrm, nrm, nrm, nrm };
            acc = __builtin_amdgcn_wmma_f32_16x16x32_f16(
                      false, ua0.v, false, ub0.v, (short)0, acc, false, false);
            acc = __builtin_amdgcn_wmma_f32_16x16x32_f16(
                      false, ua1.v, false, ub1.v, (short)0, acc, false, false);

            #pragma unroll
            for (int r = 0; r < 8; ++r) {
                float d  = acc[r];
                bool  lt = d < best[r];           // strict: keep earliest index
                best[r]  = lt ? d    : best[r];
                bidx[r]  = lt ? code : bidx[r];
            }
        }
    }

    // ---- reduce min/argmin across the 16 lane-columns of each half ----
    #pragma unroll
    for (int r = 0; r < 8; ++r) {
        float m = best[r];
        int   i = bidx[r];
        #pragma unroll
        for (int s = 1; s < 16; s <<= 1) {
            float m2 = __shfl_xor(m, s, 32);
            int   i2 = __shfl_xor(i, s, 32);
            bool take = (m2 < m) || (m2 == m && i2 < i);   // first-index tie-break
            m = take ? m2 : m;
            i = take ? i2 : i;
        }
        if (lcol == 0)                       // lane 0 -> rows r..; lane 16 -> rows r+8
            lds_idx[wave * 16 + 8 * hgrp + r] = i;
    }
    __syncthreads();

    // ---- gather winning code vectors (fp32) + emit indices ----
    #pragma unroll
    for (int j = 0; j < 8; ++j) {
        int i4   = tid + 256 * j;
        int row  = i4 >> 4;
        int c4   = i4 & 15;
        int code = lds_idx[row];
        float4 v = ((const float4*)cb)[code * (EMB / 4) + c4];
        ((float4*)out_q)[(rowbase + row) * (EMB / 4) + c4] = v;
    }
    if (tid < ROWS_PER_BLOCK)
        out_idx[rowbase + tid] = (float)lds_idx[tid];
}

extern "C" void kernel_launch(void* const* d_in, const int* in_sizes, int n_in,
                              void* d_out, int out_size, void* d_ws, size_t ws_size,
                              hipStream_t stream) {
    const float* x  = (const float*)d_in[0];   // [64*4096, 64]
    const float* cb = (const float*)d_in[1];   // [1024, 64]
    const long N = 64L * 4096L;
    float* out_q   = (float*)d_out;
    float* out_idx = (float*)d_out + N * EMB;

    int blocks = (int)(N / ROWS_PER_BLOCK);    // 2048
    hipLaunchKernelGGL(vq_argmin_kernel, dim3(blocks), dim3(256), 0, stream,
                       x, cb, out_q, out_idx);
}
